// ImpedanceAwareGCN_71811853189550
// MI455X (gfx1250) — matrix operand order repaired and checked by reading, not compiled, not run on hardware
//
#include <hip/hip_runtime.h>

typedef __bf16 bf16_t;
typedef __attribute__((ext_vector_type(16))) __bf16 v16bf;
typedef __attribute__((ext_vector_type(8)))  float  v8f;

#define NCH 64  // hidden width (H1 == H2 == OUT_DIM == 64)

static __device__ __forceinline__ bf16_t f2bf(float f) {
  // round-to-nearest-even fp32 -> bf16
  unsigned u = __builtin_bit_cast(unsigned, f);
  u += 0x7fffu + ((u >> 16) & 1u);
  unsigned short h = (unsigned short)(u >> 16);
  return __builtin_bit_cast(bf16_t, h);
}

// ---------------------------------------------------------------------------
// Dual GEMM: tout = h @ W + b ; timp = h @ Wi
// One wave per 16-row tile; 4 column tiles of 16; WMMA bf16 -> f32 accum.
// LEAKY applies leaky_relu(0.2) to the *input* h while building A fragments
// (fuses the activation of the previous layer's aggregation).
// ---------------------------------------------------------------------------
template <int K, bool LEAKY>
__global__ __launch_bounds__(256)
void gemm_dual(const float* __restrict__ h,
               const float* __restrict__ W,
               const float* __restrict__ bias,
               const float* __restrict__ Wi,
               float* __restrict__ tout,
               float* __restrict__ timp,
               int nrows)
{
  constexpr int KP = K + 16;                    // padded LDS row stride (bf16), 32B multiple
  __shared__ alignas(32) bf16_t sW [NCH * KP];  // transposed: sW[n*KP + k]
  __shared__ alignas(32) bf16_t sWi[NCH * KP];
  __shared__ float sB[NCH];

  const int tid = threadIdx.x;
  for (int i = tid; i < K * NCH; i += 256) {
    int k = i / NCH, n = i % NCH;               // W is [K][64] row-major
    sW [n * KP + k] = f2bf(W [i]);
    sWi[n * KP + k] = f2bf(Wi[i]);
  }
  if (tid < NCH) sB[tid] = bias[tid];
  __syncthreads();

  const int wave = tid >> 5;
  const int lane = tid & 31;
  const int tile = blockIdx.x * 8 + wave;
  const int ntiles = (nrows + 15) >> 4;
  if (tile >= ntiles) return;                   // wave-uniform exit (WMMA needs EXEC all-1)

  const int m    = lane & 15;
  const int half = lane >> 4;                   // 0: lanes 0-15, 1: lanes 16-31
  int row = tile * 16 + m;
  if (row >= nrows) row = nrows - 1;            // clamp (duplicate reads are harmless)
  const float* hrow = h + (size_t)row * K;

  v8f accO[4] = {};
  v8f accI[4] = {};

  #pragma unroll
  for (int kt = 0; kt < K / 32; ++kt) {
    // ---- A fragment (16x32 bf16): contiguous k order (k = 16*half + elem), same as B ----
    // lanes 0-15 : elems 0-15 -> K 0-15
    // lanes 16-31: elems 0-15 -> K 16-31
    const int kb = kt * 32 + half * 16;
    v16bf a;
    #pragma unroll
    for (int j = 0; j < 8; ++j) {
      float v0 = hrow[kb + j];
      float v1 = hrow[kb + 8 + j];
      if (LEAKY) {
        v0 = (v0 >= 0.f) ? v0 : 0.2f * v0;
        v1 = (v1 >= 0.f) ? v1 : 0.2f * v1;
      }
      a[j]     = f2bf(v0);
      a[8 + j] = f2bf(v1);
    }
    // ---- B fragments (32x16 bf16): lane n = lane&15; K block by lane half ----
    const int kcol = kt * 32 + half * 16;       // 32B-aligned within padded row
    #pragma unroll
    for (int nt = 0; nt < 4; ++nt) {
      const int n = nt * 16 + (lane & 15);
      v16bf bW = *(const v16bf*)&sW [n * KP + kcol];
      v16bf bI = *(const v16bf*)&sWi[n * KP + kcol];
      accO[nt] = __builtin_amdgcn_wmma_f32_16x16x32_bf16(
          false, a, false, bW, (short)0, accO[nt], false, false);
      accI[nt] = __builtin_amdgcn_wmma_f32_16x16x32_bf16(
          false, a, false, bI, (short)0, accI[nt], false, false);
    }
  }

  // ---- store D (16x16 f32 layout: vgpr r -> M = r + 8*half, N = lane&15) ----
  const int ncol = lane & 15;
  float bv[4];
  #pragma unroll
  for (int nt = 0; nt < 4; ++nt) bv[nt] = sB[nt * 16 + ncol];

  const size_t base = (size_t)(tile * 16 + 8 * half) * NCH + ncol;
  if (tile * 16 + 16 <= nrows) {                // common case: full tile, no guards
    #pragma unroll
    for (int nt = 0; nt < 4; ++nt) {
      #pragma unroll
      for (int r = 0; r < 8; ++r) {
        size_t idx = base + (size_t)r * NCH + nt * 16;
        tout[idx] = accO[nt][r] + bv[nt];
        timp[idx] = accI[nt][r];
      }
    }
  } else {                                      // ragged last tile
    #pragma unroll
    for (int nt = 0; nt < 4; ++nt) {
      #pragma unroll
      for (int r = 0; r < 8; ++r) {
        int grow = tile * 16 + r + 8 * half;
        if (grow < nrows) {
          size_t idx = (size_t)grow * NCH + nt * 16 + ncol;
          tout[idx] = accO[nt][r] + bv[nt];
          timp[idx] = accI[nt][r];
        }
      }
    }
  }
}

// ---------------------------------------------------------------------------
// Edge phase: msg = tout[row] + impedance[row] * timp[col]; H[col] += msg
// 16 edges per 256-thread block; indices/impedance staged once through LDS;
// 16 threads per edge, 4 channels each via float4 gathers. Gathers/atomics
// are L2-resident (working set 77MB << 192MB L2).
// ---------------------------------------------------------------------------
#define EPB 16
__global__ __launch_bounds__(256)
void edge_scatter(const long long* __restrict__ ei,   // [2, E] int64
                  const float* __restrict__ imp,      // [N, 1]
                  const float* __restrict__ tout,
                  const float* __restrict__ timp,
                  float* __restrict__ H,
                  int E)
{
  __shared__ int   sRow[EPB];
  __shared__ int   sCol[EPB];
  __shared__ float sImp[EPB];

  const int ebase = blockIdx.x * EPB;
  const int tid   = threadIdx.x;
  if (tid < EPB) {
    int e = ebase + tid;
    if (e >= E) e = E - 1;
    int r = (int)ei[e];
    sRow[tid] = r;
    sImp[tid] = imp[r];
  } else if (tid < 2 * EPB) {
    int e = ebase + tid - EPB;
    if (e >= E) e = E - 1;
    sCol[tid - EPB] = (int)ei[(size_t)E + e];
  }
  __syncthreads();

  const int el = tid >> 4;                      // edge slot 0..15
  const int e  = ebase + el;
  if (e >= E) return;
  const int c  = (tid & 15) * 4;                // channel quad
  const int r   = sRow[el];
  const int col = sCol[el];
  const float im = sImp[el];

  const float4 o = *(const float4*)&tout[(size_t)r   * NCH + c];
  const float4 q = *(const float4*)&timp[(size_t)col * NCH + c];
  float* Hp = &H[(size_t)col * NCH + c];
  atomicAdd(Hp + 0, o.x + im * q.x);
  atomicAdd(Hp + 1, o.y + im * q.y);
  atomicAdd(Hp + 2, o.z + im * q.z);
  atomicAdd(Hp + 3, o.w + im * q.w);
}

// ---------------------------------------------------------------------------
// Final FC: out[i] = h3[i,:] @ Wfc + bfc
// ---------------------------------------------------------------------------
__global__ __launch_bounds__(256)
void fc_kernel(const float* __restrict__ h,
               const float* __restrict__ Wfc,
               const float* __restrict__ bfc,
               float* __restrict__ out, int n)
{
  __shared__ float sw[NCH];
  if (threadIdx.x < NCH) sw[threadIdx.x] = Wfc[threadIdx.x];
  __syncthreads();
  int i = blockIdx.x * 256 + threadIdx.x;
  if (i >= n) return;
  const float4* hp = (const float4*)(h + (size_t)i * NCH);
  float acc = bfc[0];
  #pragma unroll
  for (int j = 0; j < NCH / 4; ++j) {
    float4 v = hp[j];
    acc += v.x * sw[4 * j] + v.y * sw[4 * j + 1] +
           v.z * sw[4 * j + 2] + v.w * sw[4 * j + 3];
  }
  out[i] = acc;
}

// ---------------------------------------------------------------------------
extern "C" void kernel_launch(void* const* d_in, const int* in_sizes, int n_in,
                              void* d_out, int out_size, void* d_ws, size_t ws_size,
                              hipStream_t stream)
{
  const float*     x   = (const float*)d_in[0];
  const long long* ei  = (const long long*)d_in[1];   // int64 edge_index [2,E]
  const float*     imp = (const float*)d_in[2];
  const float* W1  = (const float*)d_in[3];  const float* b1  = (const float*)d_in[4];
  const float* Wi1 = (const float*)d_in[5];
  const float* W2  = (const float*)d_in[6];  const float* b2  = (const float*)d_in[7];
  const float* Wi2 = (const float*)d_in[8];
  const float* W3  = (const float*)d_in[9];  const float* b3  = (const float*)d_in[10];
  const float* Wi3 = (const float*)d_in[11];
  const float* Wfc = (const float*)d_in[12]; const float* bfc = (const float*)d_in[13];

  const int N = in_sizes[0] / 32;   // x is [N, 32]
  const int E = in_sizes[1] / 2;    // edge_index is [2, E]

  const size_t NB = (size_t)N * NCH * sizeof(float);
  float* Tout = (float*)d_ws;
  float* Timp = (float*)((char*)d_ws + NB);
  float* H    = (float*)((char*)d_ws + 2 * NB);

  dim3 blk(256);
  const int gemmBlocks = (((N + 15) / 16) + 7) / 8;
  const int scatBlocks = (E + EPB - 1) / EPB;

  // Layer 1 (K=32, raw input)
  gemm_dual<32, false><<<gemmBlocks, blk, 0, stream>>>(x, W1, b1, Wi1, Tout, Timp, N);
  hipMemsetAsync(H, 0, NB, stream);
  edge_scatter<<<scatBlocks, blk, 0, stream>>>(ei, imp, Tout, Timp, H, E);

  // Layer 2 (K=64, leaky_relu fused into A-load)
  gemm_dual<64, true><<<gemmBlocks, blk, 0, stream>>>(H, W2, b2, Wi2, Tout, Timp, N);
  hipMemsetAsync(H, 0, NB, stream);
  edge_scatter<<<scatBlocks, blk, 0, stream>>>(ei, imp, Tout, Timp, H, E);

  // Layer 3 (K=64, leaky_relu fused into A-load; no activation after)
  gemm_dual<64, true><<<gemmBlocks, blk, 0, stream>>>(H, W3, b3, Wi3, Tout, Timp, N);
  hipMemsetAsync(H, 0, NB, stream);
  edge_scatter<<<scatBlocks, blk, 0, stream>>>(ei, imp, Tout, Timp, H, E);

  // Final FC -> d_out [N,1]
  fc_kernel<<<(N + 255) / 256, blk, 0, stream>>>(H, Wfc, bfc, (float*)d_out, N);
}
